// WHTConv2D_20194936226186
// MI455X (gfx1250) — compile-verified, matrix operand order checked by reading, not run on hardware
//
#include <hip/hip_runtime.h>
#include <hip/hip_bf16.h>

typedef __attribute__((ext_vector_type(16))) __bf16          v16bf;
typedef __attribute__((ext_vector_type(8)))  float           v8f;
typedef __attribute__((ext_vector_type(16))) unsigned short  ush16;
typedef __attribute__((ext_vector_type(4)))  int             v4i;

#define BSZ 32
#define CCH 64
#define HW  128
#define IMG (HW * HW) /* 16384 */

#define AS1 __attribute__((address_space(1)))
#define AS3 __attribute__((address_space(3)))

#ifndef __has_builtin
#define __has_builtin(x) 0
#endif
#if __has_builtin(__builtin_amdgcn_global_load_async_to_lds_b128)
#define HAS_ASYNC_LDS 1
#else
#define HAS_ASYNC_LDS 0
#endif

__device__ __forceinline__ void async_wait0() {
#if __has_builtin(__builtin_amdgcn_s_wait_asynccnt)
  __builtin_amdgcn_s_wait_asynccnt(0);
#else
  asm volatile("s_wait_asynccnt 0" ::: "memory");
#endif
}

__device__ __forceinline__ unsigned short f2bf(float f) {
  unsigned u = __builtin_bit_cast(unsigned, f);
  u += 0x7FFFu + ((u >> 16) & 1u);          // round-to-nearest-even
  return (unsigned short)(u >> 16);
}

// Load one 128x128 f32 image into LDS: async global->LDS if available.
__device__ __forceinline__ void load_image_lds(float* s, const float* g, int tid) {
#if HAS_ASYNC_LDS
#pragma unroll
  for (int i = 0; i < 16; ++i) {
    int e = i * 1024 + tid * 4;
    __builtin_amdgcn_global_load_async_to_lds_b128(
        (AS1 v4i*)(unsigned long long)(const void*)(g + e),
        (AS3 v4i*)(s + e), 0, 0);
  }
  async_wait0();
#else
#pragma unroll
  for (int i = 0; i < 16; ++i) {
    int e = i * 1024 + tid * 4;
    *(float4*)(s + e) = *(const float4*)(g + e);
  }
#endif
}

// In-LDS 2-D WHT over a 128x128 tile: per axis, 3 radix-4 passes + 1 radix-2.
__device__ __forceinline__ void wht2d_lds(float* s, int tid) {
  // ---- W axis ----
#pragma unroll
  for (int h = 1; h <= 16; h *= 4) {
    __syncthreads();
#pragma unroll
    for (int it = 0; it < 16; ++it) {
      int q = it * 256 + tid;                 // 4096 quads
      int row = q >> 5, t = q & 31;
      int pos = ((t & ~(h - 1)) << 2) | (t & (h - 1));
      int i0 = row * HW + pos;
      float a = s[i0], b = s[i0 + h], c = s[i0 + 2 * h], d = s[i0 + 3 * h];
      float ab = a + b, amb = a - b, cd = c + d, cmd = c - d;
      s[i0]         = ab + cd;
      s[i0 + h]     = amb + cmd;
      s[i0 + 2 * h] = ab - cd;
      s[i0 + 3 * h] = amb - cmd;
    }
  }
  __syncthreads();
#pragma unroll
  for (int it = 0; it < 32; ++it) {           // radix-2, h = 64
    int idx = it * 256 + tid;
    int row = idx >> 6, j = idx & 63;
    int i0 = row * HW + j, i1 = i0 + 64;
    float a = s[i0], b = s[i1];
    s[i0] = a + b; s[i1] = a - b;
  }
  // ---- H axis ----
#pragma unroll
  for (int h = 1; h <= 16; h *= 4) {
    __syncthreads();
#pragma unroll
    for (int it = 0; it < 16; ++it) {
      int q = it * 256 + tid;
      int col = q & 127, t = q >> 7;
      int pos = ((t & ~(h - 1)) << 2) | (t & (h - 1));
      int i0 = pos * HW + col, st = h * HW;
      float a = s[i0], b = s[i0 + st], c = s[i0 + 2 * st], d = s[i0 + 3 * st];
      float ab = a + b, amb = a - b, cd = c + d, cmd = c - d;
      s[i0]          = ab + cd;
      s[i0 + st]     = amb + cmd;
      s[i0 + 2 * st] = ab - cd;
      s[i0 + 3 * st] = amb - cmd;
    }
  }
  __syncthreads();
#pragma unroll
  for (int it = 0; it < 32; ++it) {           // radix-2, h = 64
    int idx = it * 256 + tid;
    int col = idx & 127, j = idx >> 7;
    int i0 = j * HW + col, i1 = i0 + 64 * HW;
    float a = s[i0], b = s[i1];
    s[i0] = a + b; s[i1] = a - b;
  }
}

// ---------------------------------------------------------------------------
// Kernel 1: forward 2-D WHT per (b,c) image; output bf16 into workspace.
// ---------------------------------------------------------------------------
__global__ void __launch_bounds__(256) k_wht_fwd(const float* __restrict__ x,
                                                 unsigned short* __restrict__ f2) {
  __shared__ float s[IMG];                    // 64 KB
  const int  tid  = threadIdx.x;
  const long base = (long)blockIdx.x * IMG;

  load_image_lds(s, x + base, tid);
  wht2d_lds(s, tid);
  __syncthreads();
#pragma unroll
  for (int i = 0; i < 16; ++i) {
    int e = i * 1024 + tid * 4;
    ushort4 o;
    o.x = f2bf(s[e + 0]); o.y = f2bf(s[e + 1]);
    o.z = f2bf(s[e + 2]); o.w = f2bf(s[e + 3]);
    *(ushort4*)(f2 + base + e) = o;
  }
}

// ---------------------------------------------------------------------------
// Kernel 2: per-pixel channel GEMM via bf16 WMMA, both pods in one pass.
//   f4_i = v_i ⊙ (W_i @ f2)   (v is channel-independent -> commutes with GEMM)
//   f6   = soft_thr(f4_0,T_0) + soft_thr(f4_1,T_1)  -> written into d_out
// One WG per (b, h-row). 8 waves = 4 co-strips x 2 N-halves.
// A and B fragments both staged through LDS in WMMA lane order.
// ---------------------------------------------------------------------------
__global__ void __launch_bounds__(256) k_pod_gemm(const unsigned short* __restrict__ f2,
                                                  const float* __restrict__ v,
                                                  const float* __restrict__ T,
                                                  const float* __restrict__ cw,
                                                  float* __restrict__ f6) {
  __shared__ unsigned short sB[8 * 2 * 32 * 16];      // [nt][kc][lane][j]      16 KB
  __shared__ unsigned short sA[4 * 2 * 2 * 32 * 16];  // [strip][pod][kc][lane][j] 16 KB
  const int tid  = threadIdx.x;
  const int b    = blockIdx.x >> 7;
  const int hrow = blockIdx.x & 127;

  // ---- stage B tile: K=64 channels x N=128 pixels (bf16), swizzled ----
  {
    const int c  = tid >> 2;       // channel (K)
    const int pb = tid & 3;        // 32-pixel block
    const long gbase = (long)(b * CCH + c) * IMG + (long)hrow * HW + pb * 32;
    const int kc = c >> 5;
    const int k  = c & 31;
    const int hi = ((k & 15) >= 8) ? 1 : 0;
    const int j0 = (k & 7) + ((k >= 16) ? 8 : 0);
#pragma unroll
    for (int p8 = 0; p8 < 8; ++p8) {
      ushort4 val = *(const ushort4*)(f2 + gbase + p8 * 4);
      unsigned short e[4] = {val.x, val.y, val.z, val.w};
#pragma unroll
      for (int q = 0; q < 4; ++q) {
        int pix = pb * 32 + p8 * 4 + q;
        int nt = pix >> 4, n = pix & 15;
        sB[(((nt * 2 + kc) * 32) + (hi * 16 + n)) * 16 + j0] = e[q];
      }
    }
  }

  // ---- stage A: conv_w (2 pods x 64x64 f32) -> bf16 in A-fragment order ----
  {
    const int fbase = tid * 32;    // 8192 weights total, 32 per thread
#pragma unroll
    for (int ql = 0; ql < 8; ++ql) {
      float4 wv = *(const float4*)(cw + fbase + ql * 4);
      float e[4] = {wv.x, wv.y, wv.z, wv.w};
#pragma unroll
      for (int q = 0; q < 4; ++q) {
        int flat = fbase + ql * 4 + q;        // = (pod*64 + co)*64 + cin
        int pod = flat >> 12;
        int co  = (flat >> 6) & 63;
        int cin = flat & 63;
        int strip = co >> 4, m = co & 15;
        int kc = cin >> 5, k = cin & 31;
        int hi = ((k & 15) >= 8) ? 1 : 0;     // ISA A layout: lanes 16-31 hold K 8-15/24-31
        int j  = (k & 7) + ((k >= 16) ? 8 : 0);
        sA[((((strip * 2 + pod) * 2 + kc) * 32) + (m + 16 * hi)) * 16 + j] = f2bf(e[q]);
      }
    }
  }
  __syncthreads();

  const int lane    = tid & 31;
  const int wave    = tid >> 5;
  const int costrip = wave & 3;
  const int nhalf   = wave >> 2;
  const int ahi     = lane >> 4;

  // ---- A fragments from LDS (2 x ds_load_b128 each) ----
  v16bf A[2][2];
#pragma unroll
  for (int pod = 0; pod < 2; ++pod)
#pragma unroll
    for (int kc = 0; kc < 2; ++kc) {
      ush16 raw = *(const ush16*)(sA + ((((costrip * 2 + pod) * 2 + kc) * 32) + lane) * 16);
      A[pod][kc] = __builtin_bit_cast(v16bf, raw);
    }

  // ---- GEMM: 16 v_wmma_f32_16x16x32_bf16 per wave ----
  v8f acc[2][4];
#pragma unroll
  for (int p = 0; p < 2; ++p)
#pragma unroll
    for (int t = 0; t < 4; ++t)
      acc[p][t] = (v8f){0.f, 0.f, 0.f, 0.f, 0.f, 0.f, 0.f, 0.f};

#pragma unroll
  for (int ntl = 0; ntl < 4; ++ntl) {
    const int nt = nhalf * 4 + ntl;
#pragma unroll
    for (int kc = 0; kc < 2; ++kc) {
      ush16 raw = *(const ush16*)(sB + (((nt * 2 + kc) * 32) + lane) * 16);
      v16bf Bf  = __builtin_bit_cast(v16bf, raw);
      acc[0][ntl] = __builtin_amdgcn_wmma_f32_16x16x32_bf16(
          false, A[0][kc], false, Bf, (short)0, acc[0][ntl], false, false);
      acc[1][ntl] = __builtin_amdgcn_wmma_f32_16x16x32_bf16(
          false, A[1][kc], false, Bf, (short)0, acc[1][ntl], false, false);
    }
  }

  // ---- epilogue: v-scale, soft-threshold, pod sum, store f32 ----
#pragma unroll
  for (int ntl = 0; ntl < 4; ++ntl) {
    const int nt   = nhalf * 4 + ntl;
    const int wcol = nt * 16 + (lane & 15);
    const int pixg = hrow * HW + wcol;
    const float v0  = v[pixg];
    const float v1  = v[IMG + pixg];
    const float th0 = fmaxf(T[pixg], 0.f);
    const float th1 = fmaxf(T[IMG + pixg], 0.f);
#pragma unroll
    for (int r = 0; r < 8; ++r) {
      const int co = costrip * 16 + r + 8 * ahi; // D layout: lanes>=16 -> M=r+8
      float a0 = v0 * acc[0][ntl][r];
      float a1 = v1 * acc[1][ntl][r];
      float s0 = copysignf(fmaxf(fabsf(a0) - th0, 0.f), a0);
      float s1 = copysignf(fmaxf(fabsf(a1) - th1, 0.f), a1);
      f6[(long)(b * CCH + co) * IMG + pixg] = s0 + s1;
    }
  }
}

// ---------------------------------------------------------------------------
// Kernel 3: inverse 2-D WHT of f6 (in d_out, in-place), *1/16384, + x.
// ---------------------------------------------------------------------------
__global__ void __launch_bounds__(256) k_wht_inv_add(const float* __restrict__ x,
                                                     float* __restrict__ out) {
  __shared__ float s[IMG];
  const int  tid  = threadIdx.x;
  const long base = (long)blockIdx.x * IMG;

  load_image_lds(s, out + base, tid);
  wht2d_lds(s, tid);
  __syncthreads();
  const float inv = 1.0f / 16384.0f;
#pragma unroll
  for (int i = 0; i < 16; ++i) {
    int e = i * 1024 + tid * 4;
    float4 xa = *(const float4*)(x + base + e);
    float4 o;
    o.x = s[e + 0] * inv + xa.x;
    o.y = s[e + 1] * inv + xa.y;
    o.z = s[e + 2] * inv + xa.z;
    o.w = s[e + 3] * inv + xa.w;
    *(float4*)(out + base + e) = o;
  }
}

extern "C" void kernel_launch(void* const* d_in, const int* in_sizes, int n_in,
                              void* d_out, int out_size, void* d_ws, size_t ws_size,
                              hipStream_t stream) {
  const float* x  = (const float*)d_in[0];   // (B, C_IN, H, W)
  const float* v  = (const float*)d_in[1];   // (PODS, H, W)
  const float* T  = (const float*)d_in[2];   // (PODS, H, W)
  const float* cw = (const float*)d_in[3];   // (PODS, C_OUT, C_IN)
  float* out = (float*)d_out;
  unsigned short* f2 = (unsigned short*)d_ws; // bf16, B*C_IN*H*W = 64 MiB

  // 1) forward 2-D WHT: one WG per (b, c) image
  k_wht_fwd<<<BSZ * CCH, 256, 0, stream>>>(x, f2);
  // 2) fused pod GEMM + epilogue: one WG per (b, h-row); f6 -> d_out
  k_pod_gemm<<<BSZ * HW, 256, 0, stream>>>(f2, v, T, cw, out);
  // 3) inverse 2-D WHT + residual add, in-place on d_out
  k_wht_inv_add<<<BSZ * CCH, 256, 0, stream>>>(x, out);
}
